// CharBiLSTMEmbedder_22050362098248
// MI455X (gfx1250) — compile-verified, matrix operand order checked by reading, not run on hardware
//
#include <hip/hip_runtime.h>
#include <hip/hip_bf16.h>
#include <math.h>

typedef __attribute__((ext_vector_type(16))) _Float16 v16h;
typedef __attribute__((ext_vector_type(8)))  float    v8f;

#define TT     20      // timesteps
#define EE     50      // embedding dim
#define HH     50      // hidden dim
#define GG     200     // 4*H gate dim
#define KZ     128     // padded K (E+H=100 -> 128)
#define NTILES 13      // ceil(200/16)
#define KTILES 4       // 128/32
#define NPAD   208     // NTILES*16
#define PADIDX 0

// Hardware tanh (TRANS32 on CDNA5). Builtin if declared, else inline asm with
// a trailing v_nop to satisfy the 1-op TRANS result-use hazard (ISA 7.4).
#if defined(__has_builtin) && __has_builtin(__builtin_amdgcn_tanhf)
__device__ __forceinline__ float fast_tanh(float x) { return __builtin_amdgcn_tanhf(x); }
#else
__device__ __forceinline__ float fast_tanh(float x) {
    float y;
    asm volatile("v_tanh_f32 %0, %1\n\tv_nop\n\tv_nop" : "=v"(y) : "v"(x));
    return y;
}
#endif

__device__ __forceinline__ float fast_sigmoid(float x) {
    // sigmoid(x) = 0.5 + 0.5*tanh(x/2)
    return fmaf(0.5f, fast_tanh(0.5f * x), 0.5f);
}

// ---------------------------------------------------------------------------
// Prep kernel: swizzle [W_ih | W_hh]^T (padded to 128x208, f16) for both
// directions into the exact per-lane WMMA B-fragment layout, and fuse biases.
// B-fragment layout (16-bit, 16x16x32): lane<16 -> col N=lane, K in
// {0..7,16..23}; lane>=16 -> col N=lane-16, K in {8..15,24..31}; packed as
// f16 pairs per dword (VGPR v<4: K=2v,2v+1 (+koff); v>=4: K=16+2(v-4),...).
// ---------------------------------------------------------------------------
__global__ void bilstm_prep_kernel(
    const float* __restrict__ Wih_f, const float* __restrict__ Whh_f,
    const float* __restrict__ bih_f, const float* __restrict__ bhh_f,
    const float* __restrict__ Wih_b, const float* __restrict__ Whh_b,
    const float* __restrict__ bih_b, const float* __restrict__ bhh_b,
    _Float16* __restrict__ Bws, float* __restrict__ biasws)
{
    const int tid = blockIdx.x * blockDim.x + threadIdx.x;
    const int NFRAGLANE = 2 * KTILES * NTILES * 32;   // 3328
    if (tid < NFRAGLANE) {
        int e    = tid;
        int lane = e & 31;  e >>= 5;
        int nt   = e % NTILES; e /= NTILES;
        int kt   = e & 3;   e >>= 2;
        int dir  = e;
        const float* Wih = dir ? Wih_b : Wih_f;
        const float* Whh = dir ? Whh_b : Whh_f;
        const int n     = nt * 16 + (lane & 15);
        const int kbase = kt * 32 + ((lane >> 4) ? 8 : 0);
        _Float16* dst = Bws + (size_t)((((dir * KTILES) + kt) * NTILES + nt) * 32 + lane) * 16;
        #pragma unroll
        for (int v = 0; v < 8; ++v) {
            const int k0 = kbase + ((v < 4) ? 2 * v : 16 + 2 * (v - 4));
            #pragma unroll
            for (int p = 0; p < 2; ++p) {
                const int k = k0 + p;
                float w = 0.0f;
                if (n < GG) {
                    if (k < EE)            w = Wih[n * EE + k];
                    else if (k < EE + HH)  w = Whh[n * HH + (k - EE)];
                }
                dst[v * 2 + p] = (_Float16)w;
            }
        }
    } else if (tid < NFRAGLANE + 2 * GG) {
        int q   = tid - NFRAGLANE;
        int dir = q / GG;
        int j   = q % GG;
        biasws[dir * GG + j] = dir ? (bih_b[j] + bhh_b[j]) : (bih_f[j] + bhh_f[j]);
    }
}

// ---------------------------------------------------------------------------
// Main kernel: one wave32 per 16-word tile. Fused GEMM per step:
// z[16x128] (f16: cols 0..49 = x_t, 50..99 = h, rest 0) x W^T[128x208]
// via 13x4 v_wmma_f32_16x16x32_f16 (bias pre-seeded in C), then masked
// LSTM cell update using hardware v_tanh_f32.
// ---------------------------------------------------------------------------
__global__ __launch_bounds__(32) void bilstm_main_kernel(
    const int*   __restrict__ cidx,   // [N,T]
    const int*   __restrict__ wlen,   // [N]
    const float* __restrict__ emb,    // [V,E]
    const _Float16* __restrict__ Bws, // swizzled fragments
    const float* __restrict__ biasws, // [2,200]
    float* __restrict__ out,          // [N, 2H]
    int Nw)
{
    __shared__ _Float16 zbuf[16][KZ];     // 4 KB
    __shared__ float    gbuf[16][NPAD];   // 13 KB
    __shared__ float    cbuf[16][HH];     // 3.2 KB
    __shared__ float    hbuf[16][HH];     // 3.2 KB
    __shared__ int      ilen[16];
    __shared__ int      idxt[16];

    const int lane = threadIdx.x;
    const int w0   = blockIdx.x * 16;

    if (lane < 16) {
        const int w = w0 + lane;
        ilen[lane] = (w < Nw) ? wlen[w] : 0;   // len=0 -> h stays 0
    }
    for (int i = lane; i < 16 * KZ; i += 32)
        ((_Float16*)zbuf)[i] = (_Float16)0.0f;
    __syncthreads();

    for (int dir = 0; dir < 2; ++dir) {
        // reset recurrent state
        for (int i = lane; i < 16 * HH; i += 32) {
            ((float*)cbuf)[i] = 0.0f;
            ((float*)hbuf)[i] = 0.0f;
            int m = i / HH, j = i % HH;
            zbuf[m][EE + j] = (_Float16)0.0f;
        }

        // per-lane bias for the 13 N-tiles: C/D column for this lane is
        // nt*16 + (lane&15), identical across the 8 accumulator rows.
        float bcol[NTILES];
        #pragma unroll
        for (int nt = 0; nt < NTILES; ++nt) {
            const int col = nt * 16 + (lane & 15);
            bcol[nt] = (col < GG) ? biasws[dir * GG + col] : 0.0f;
        }
        __syncthreads();

        for (int s = 0; s < TT; ++s) {
            const int t = dir ? (TT - 1 - s) : s;

            if (lane < 16) {
                const int w = w0 + lane;
                idxt[lane] = (w < Nw) ? cidx[w * TT + t] : PADIDX;
            }
            __syncthreads();

            // embedding gather -> z x-columns (f16); PAD row forced to zero
            for (int i = lane; i < 16 * EE; i += 32) {
                const int m = i / EE, k = i % EE;
                const int idx = idxt[m];
                const float xv = (idx == PADIDX) ? 0.0f : emb[idx * EE + k];
                zbuf[m][k] = (_Float16)xv;
            }
            __syncthreads();

            // A fragments from LDS z (16-bit A layout)
            v16h a[KTILES];
            {
                const int m    = lane & 15;
                const int koff = (lane >> 4) * 8;
                #pragma unroll
                for (int kt = 0; kt < KTILES; ++kt) {
                    #pragma unroll
                    for (int v = 0; v < 8; ++v) {
                        const int k0 = kt * 32 + koff + ((v < 4) ? 2 * v : 16 + 2 * (v - 4));
                        a[kt][2 * v]     = zbuf[m][k0];
                        a[kt][2 * v + 1] = zbuf[m][k0 + 1];
                    }
                }
            }

            // GEMM: 13 N-tiles x 4 K-tiles of v_wmma_f32_16x16x32_f16,
            // bias pre-seeded into the accumulator.
            const v16h* Bfrag = (const v16h*)Bws;
            #pragma unroll
            for (int nt = 0; nt < NTILES; ++nt) {
                v8f acc;
                #pragma unroll
                for (int r = 0; r < 8; ++r) acc[r] = bcol[nt];
                #pragma unroll
                for (int kt = 0; kt < KTILES; ++kt) {
                    const v16h b = Bfrag[(size_t)(((dir * KTILES) + kt) * NTILES + nt) * 32 + lane];
                    acc = __builtin_amdgcn_wmma_f32_16x16x32_f16(
                        false, a[kt], false, b, (short)0, acc, false, false);
                }
                // C/D layout: lane<16 -> (M=r, N=lane); lane>=16 -> (M=8+r, N=lane-16)
                const int mrow = (lane >> 4) * 8;
                const int col  = nt * 16 + (lane & 15);
                #pragma unroll
                for (int r = 0; r < 8; ++r)
                    gbuf[mrow + r][col] = acc[r];
            }
            __syncthreads();

            // LSTM cell update, masked by t < len; hardware tanh path
            for (int i = lane; i < 16 * HH; i += 32) {
                const int m = i / HH, j = i % HH;
                if (t < ilen[m]) {
                    const float iv = gbuf[m][j];
                    const float fv = gbuf[m][HH + j];
                    const float gv = gbuf[m][2 * HH + j];
                    const float ov = gbuf[m][3 * HH + j];
                    const float si = fast_sigmoid(iv);
                    const float sf = fast_sigmoid(fv);
                    const float so = fast_sigmoid(ov);
                    const float cn = sf * cbuf[m][j] + si * fast_tanh(gv);
                    const float hn = so * fast_tanh(cn);
                    cbuf[m][j] = cn;
                    hbuf[m][j] = hn;
                    zbuf[m][EE + j] = (_Float16)hn;   // feed next step's WMMA
                }
            }
            __syncthreads();
        }

        // write final hidden state for this direction
        for (int i = lane; i < 16 * HH; i += 32) {
            const int m = i / HH, j = i % HH;
            const int w = w0 + m;
            if (w < Nw)
                out[(size_t)w * (2 * HH) + dir * HH + j] = hbuf[m][j];
        }
        __syncthreads();
    }
}

extern "C" void kernel_launch(void* const* d_in, const int* in_sizes, int n_in,
                              void* d_out, int out_size, void* d_ws, size_t ws_size,
                              hipStream_t stream) {
    const int*   cidx  = (const int*)  d_in[0];
    const int*   wlen  = (const int*)  d_in[1];
    const float* emb   = (const float*)d_in[2];
    const float* Wih_f = (const float*)d_in[3];
    const float* Whh_f = (const float*)d_in[4];
    const float* bih_f = (const float*)d_in[5];
    const float* bhh_f = (const float*)d_in[6];
    const float* Wih_b = (const float*)d_in[7];
    const float* Whh_b = (const float*)d_in[8];
    const float* bih_b = (const float*)d_in[9];
    const float* bhh_b = (const float*)d_in[10];
    float* out = (float*)d_out;

    const int Nw = in_sizes[1];  // word count

    // workspace layout: [B fragments: 2*4*13*32 lanes * 16 f16 = 106496 B][bias: 2*200 f32]
    _Float16* Bws    = (_Float16*)d_ws;
    float*    biasws = (float*)((char*)d_ws + (size_t)2 * KTILES * NTILES * 32 * 16 * sizeof(_Float16));

    const int prep_work = 2 * KTILES * NTILES * 32 + 2 * GG;
    bilstm_prep_kernel<<<(prep_work + 255) / 256, 256, 0, stream>>>(
        Wih_f, Whh_f, bih_f, bhh_f, Wih_b, Whh_b, bih_b, bhh_b, Bws, biasws);

    const int nblk = (Nw + 15) / 16;
    bilstm_main_kernel<<<nblk, 32, 0, stream>>>(
        cidx, wlen, emb, Bws, biasws, out, Nw);
}
